// InterestEvolving_72791105733094
// MI455X (gfx1250) — matrix operand order repaired and checked by reading, not compiled, not run on hardware
//
#include <hip/hip_runtime.h>

typedef __attribute__((ext_vector_type(16))) _Float16 v16h;
typedef __attribute__((ext_vector_type(8)))  float    v8f;

#define B_  2048
#define T_  200
#define H_  128
#define A1_ 64
#define A2_ 16

// Dynamic-LDS layout for the AUGRU kernel (320KB/WGP on CDNA5 makes this legal)
#define HROW_PITCH      (H_ + 4)
#define HBUF_BYTES      (16 * HROW_PITCH * 4)             // 8448
#define WIH_LDS_OFF     0
#define WHH_LDS_OFF     (3 * H_ * H_ * 2)                 // 98304
#define HBUF0_LDS_OFF   (2 * 3 * H_ * H_ * 2)             // 196608
#define HBUF1_LDS_OFF   (HBUF0_LDS_OFF + HBUF_BYTES)      // 205056
#define AUGRU_LDS_BYTES (HBUF1_LDS_OFF + HBUF_BYTES)      // 213504

__device__ __forceinline__ float sigmoidf_(float x) { return 1.0f / (1.0f + __expf(-x)); }

// ---- WMMA operand builders -------------------------------------------------
// A-frag (16x32 f16, M=lane&15): lane<16 holds K {kb..kb+7, kb+16..kb+23},
// lane>=16 holds K {kb+8.., kb+24..}, kb = ks*32. hi = lane>>4.
__device__ __forceinline__ v16h build_afrag(const float* row, int ks, int hi) {
    const int k0 = ks * 32 + hi * 8;
    float4 a = *(const float4*)(row + k0);
    float4 b = *(const float4*)(row + k0 + 4);
    float4 c = *(const float4*)(row + k0 + 16);
    float4 d = *(const float4*)(row + k0 + 20);
    v16h r;
    r[0]=(_Float16)a.x;  r[1]=(_Float16)a.y;  r[2]=(_Float16)a.z;  r[3]=(_Float16)a.w;
    r[4]=(_Float16)b.x;  r[5]=(_Float16)b.y;  r[6]=(_Float16)b.z;  r[7]=(_Float16)b.w;
    r[8]=(_Float16)c.x;  r[9]=(_Float16)c.y;  r[10]=(_Float16)c.z; r[11]=(_Float16)c.w;
    r[12]=(_Float16)d.x; r[13]=(_Float16)d.y; r[14]=(_Float16)d.z; r[15]=(_Float16)d.w;
    return r;
}

// Convert 4 prefetched float4 blocks (k0, k0+4, k0+16, k0+20) into an A-frag
__device__ __forceinline__ v16h cvt_frag(const float4* q) {
    v16h r;
    r[0]=(_Float16)q[0].x;  r[1]=(_Float16)q[0].y;  r[2]=(_Float16)q[0].z;  r[3]=(_Float16)q[0].w;
    r[4]=(_Float16)q[1].x;  r[5]=(_Float16)q[1].y;  r[6]=(_Float16)q[1].z;  r[7]=(_Float16)q[1].w;
    r[8]=(_Float16)q[2].x;  r[9]=(_Float16)q[2].y;  r[10]=(_Float16)q[2].z; r[11]=(_Float16)q[2].w;
    r[12]=(_Float16)q[3].x; r[13]=(_Float16)q[3].y; r[14]=(_Float16)q[3].z; r[15]=(_Float16)q[3].w;
    return r;
}

// A-frag of element-wise product of two rows (q*k feature)
__device__ __forceinline__ v16h build_afrag_prod(const float* krow, const float* qrw, int ks, int hi) {
    const int k0 = ks * 32 + hi * 8;
    float4 ka = *(const float4*)(krow + k0);
    float4 kb = *(const float4*)(krow + k0 + 4);
    float4 kc = *(const float4*)(krow + k0 + 16);
    float4 kd = *(const float4*)(krow + k0 + 20);
    float4 qa = *(const float4*)(qrw + k0);
    float4 qb = *(const float4*)(qrw + k0 + 4);
    float4 qc = *(const float4*)(qrw + k0 + 16);
    float4 qd = *(const float4*)(qrw + k0 + 20);
    v16h r;
    r[0]=(_Float16)(ka.x*qa.x);  r[1]=(_Float16)(ka.y*qa.y);  r[2]=(_Float16)(ka.z*qa.z);  r[3]=(_Float16)(ka.w*qa.w);
    r[4]=(_Float16)(kb.x*qb.x);  r[5]=(_Float16)(kb.y*qb.y);  r[6]=(_Float16)(kb.z*qb.z);  r[7]=(_Float16)(kb.w*qb.w);
    r[8]=(_Float16)(kc.x*qc.x);  r[9]=(_Float16)(kc.y*qc.y);  r[10]=(_Float16)(kc.z*qc.z); r[11]=(_Float16)(kc.w*qc.w);
    r[12]=(_Float16)(kd.x*qd.x); r[13]=(_Float16)(kd.y*qd.y); r[14]=(_Float16)(kd.z*qd.z); r[15]=(_Float16)(kd.w*qd.w);
    return r;
}

// B-frag (32x16 f16) from f16 weight stored [N, K=128] row-major.
// Per-lane: n = tile_n + (lane&15), kb = ks*32 + hi*8 -> K {kb..kb+7, kb+16..kb+23}.
__device__ __forceinline__ v16h load_bfrag(const _Float16* W, int n, int kb) {
    union { float4 f[2]; v16h h; } u;
    const _Float16* p = W + n * H_ + kb;
    u.f[0] = *(const float4*)(p);
    u.f[1] = *(const float4*)(p + 16);
    return u.h;
}

// ---- prep kernels ----------------------------------------------------------
__global__ void prep_qrow_kernel(const float* __restrict__ query,
                                 const float* __restrict__ W1,
                                 const float* __restrict__ b1,
                                 float* __restrict__ qrow) {
    int idx = blockIdx.x * blockDim.x + threadIdx.x;
    if (idx >= B_ * A1_) return;
    int b = idx / A1_, j = idx - b * A1_;
    float s = b1[j];
    const float* q = query + b * H_;
    for (int k = 0; k < H_; ++k)
        s += q[k] * (W1[k * A1_ + j] + W1[(2 * H_ + k) * A1_ + j]);
    qrow[idx] = s;
}

__global__ void prep_w16_kernel(const float* __restrict__ W1,
                                const float* __restrict__ W_ih,
                                const float* __restrict__ W_hh,
                                _Float16* __restrict__ Wk16,
                                _Float16* __restrict__ Wp16,
                                _Float16* __restrict__ Wih16,
                                _Float16* __restrict__ Whh16) {
    int idx = blockIdx.x * blockDim.x + threadIdx.x;
    if (idx < A1_ * H_) {               // 64 x 128, [n][k]
        int n = idx / H_, k = idx - n * H_;
        Wk16[idx] = (_Float16)(W1[(H_ + k) * A1_ + n] - W1[(2 * H_ + k) * A1_ + n]);
        Wp16[idx] = (_Float16)(W1[(3 * H_ + k) * A1_ + n]);
    }
    if (idx < 3 * H_ * H_) {            // 384 x 128 (already [N,K] row-major)
        Wih16[idx] = (_Float16)W_ih[idx];
        Whh16[idx] = (_Float16)W_hh[idx];
    }
}

// ---- attention scores: 4 waves/block, one 16-row (b,t)-tile per wave -------
__global__ __launch_bounds__(128)
void attn_scores_kernel(const float* __restrict__ query,
                        const float* __restrict__ keys,
                        const int*   __restrict__ klen,
                        const float* __restrict__ qrow,
                        const _Float16* __restrict__ Wk16,
                        const _Float16* __restrict__ Wp16,
                        const float* __restrict__ W2, const float* __restrict__ b2,
                        const float* __restrict__ W3, const float* __restrict__ b3,
                        float* __restrict__ scores) {
    __shared__ float h1s[4][16][A1_ + 1];
    const int lane = threadIdx.x & 31;
    const int wv   = threadIdx.x >> 5;
    const int tile = blockIdx.x * 4 + wv;
    const int li   = lane & 15;
    const int hi   = lane >> 4;
    const int mrow = tile * 16 + li;           // flattened b*T+t
    const int brow = mrow / T_;
    const float* krow = keys + (size_t)mrow * H_;
    const float* qr   = query + brow * H_;

    v8f acc[4] = {};
#pragma unroll
    for (int ks = 0; ks < 4; ++ks) {
        v16h Ak = build_afrag(krow, ks, hi);
        v16h Aq = build_afrag_prod(krow, qr, ks, hi);
        const int kb = ks * 32 + hi * 8;
#pragma unroll
        for (int nt = 0; nt < 4; ++nt) {
            v16h Bk = load_bfrag(Wk16, nt * 16 + li, kb);
            acc[nt] = __builtin_amdgcn_wmma_f32_16x16x32_f16(false, Ak, false, Bk,
                                                             (short)0, acc[nt], false, false);
            v16h Bp = load_bfrag(Wp16, nt * 16 + li, kb);
            acc[nt] = __builtin_amdgcn_wmma_f32_16x16x32_f16(false, Aq, false, Bp,
                                                             (short)0, acc[nt], false, false);
        }
    }
#pragma unroll
    for (int nt = 0; nt < 4; ++nt) {
        int n = nt * 16 + li;
#pragma unroll
        for (int i = 0; i < 8; ++i) {
            int M = i + hi * 8;
            int m = tile * 16 + M;
            int b = m / T_;
            h1s[wv][M][n] = sigmoidf_(acc[nt][i] + qrow[b * A1_ + n]);
        }
    }
    __syncthreads();
    if (lane < 16) {                            // layers 2+3 (3% of FLOPs) on VALU
        int m = tile * 16 + lane;
        int b = m / T_;
        int t = m - b * T_;
        float h2[A2_];
#pragma unroll
        for (int j = 0; j < A2_; ++j) h2[j] = b2[j];
        for (int c = 0; c < A1_; ++c) {
            float hv = h1s[wv][lane][c];
#pragma unroll
            for (int j = 0; j < A2_; ++j) h2[j] += hv * W2[c * A2_ + j];
        }
        float s = b3[0];
#pragma unroll
        for (int j = 0; j < A2_; ++j) s += sigmoidf_(h2[j]) * W3[j];
        scores[m] = (t < klen[b]) ? s : 0.0f;
    }
}

// ---- AUGRU: 16 batch rows / block, 8 waves x 16 h-columns each -------------
// Weights staged in LDS once (192KB, CDNA5 320KB WGP LDS); keys/scores for
// step t+1 prefetched into registers during step t; fp32 h double-buffered in
// LDS so the recurrence needs only ONE barrier per step.
__global__ __launch_bounds__(256)
void augru_kernel(const float* __restrict__ keys,
                  const int*   __restrict__ klen,
                  const float* __restrict__ scores,
                  const _Float16* __restrict__ Wih16g,
                  const _Float16* __restrict__ Whh16g,
                  const float* __restrict__ b_ih,
                  const float* __restrict__ b_hh,
                  float* __restrict__ out) {
    extern __shared__ char smem[];
    _Float16* WihL = (_Float16*)(smem + WIH_LDS_OFF);
    _Float16* WhhL = (_Float16*)(smem + WHH_LDS_OFF);
    float*    hb0  = (float*)(smem + HBUF0_LDS_OFF);     // [16][HROW_PITCH]
    float*    hb1  = (float*)(smem + HBUF1_LDS_OFF);     // [16][HROW_PITCH]

    const int tid  = threadIdx.x;
    const int lane = tid & 31;
    const int wave = tid >> 5;
    const int li   = lane & 15;
    const int hi   = lane >> 4;
    const int rowBase = blockIdx.x * 16;                 // batch rows
    const int colBase = wave * 16;                       // this wave's h-columns

    // one-time: stage weights into LDS (f16, [N,K] row-major), zero h buffers
    {
        const float4* s1 = (const float4*)Wih16g;
        const float4* s2 = (const float4*)Whh16g;
        float4* d1 = (float4*)WihL;
        float4* d2 = (float4*)WhhL;
        for (int i = tid; i < (3 * H_ * H_) / 8; i += 256) { d1[i] = s1[i]; d2[i] = s2[i]; }
        for (int i = tid; i < 16 * HROW_PITCH; i += 256) { hb0[i] = 0.0f; hb1[i] = 0.0f; }
    }

    int lenr[8];
#pragma unroll
    for (int i = 0; i < 8; ++i) lenr[i] = klen[rowBase + hi * 8 + i];

    const int nr = 0 * H_ + colBase + li;
    const int nz = 1 * H_ + colBase + li;
    const int nn = 2 * H_ + colBase + li;
    const float bs_r = b_ih[nr] + b_hh[nr];
    const float bs_z = b_ih[nz] + b_hh[nz];
    const float bi_n = b_ih[nn];
    const float bh_n = b_hh[nn];

    v8f hreg = {};                                       // C-layout slice of h
    const float* keyRow = keys + (size_t)(rowBase + li) * T_ * H_;

    // software pipeline: keys + att for step t prefetched during step t-1
    float4 kbuf[16];
    float  attbuf[8];
    auto prefetch = [&](int t) {
        const float* kr = keyRow + t * H_;
#pragma unroll
        for (int ks = 0; ks < 4; ++ks) {
            const int k0 = ks * 32 + hi * 8;
            kbuf[ks * 4 + 0] = *(const float4*)(kr + k0);
            kbuf[ks * 4 + 1] = *(const float4*)(kr + k0 + 4);
            kbuf[ks * 4 + 2] = *(const float4*)(kr + k0 + 16);
            kbuf[ks * 4 + 3] = *(const float4*)(kr + k0 + 20);
        }
#pragma unroll
        for (int i = 0; i < 8; ++i)
            attbuf[i] = scores[(rowBase + hi * 8 + i) * T_ + t];
    };
    prefetch(0);
    __syncthreads();                                     // weights + h ready

    for (int t = 0; t < T_; ++t) {
        // double buffer: h(t-1) lives in buf[(t+1)&1]; h(t) written to buf[t&1]
        const float* hrd = (t & 1) ? hb0 : hb1;
        float*       hwr = (t & 1) ? hb1 : hb0;

        // consume prefetched keys/att, immediately issue loads for t+1
        v16h Ak[4];
#pragma unroll
        for (int ks = 0; ks < 4; ++ks) Ak[ks] = cvt_frag(&kbuf[ks * 4]);
        float attv[8];
#pragma unroll
        for (int i = 0; i < 8; ++i) attv[i] = attbuf[i];
        prefetch(t + 1 < T_ ? t + 1 : t);                // overlaps whole step

        // h A-frags from LDS fp32 master copy (read buffer only)
        const float* hrow = hrd + li * HROW_PITCH;
        v16h Ah[4];
#pragma unroll
        for (int ks = 0; ks < 4; ++ks) Ah[ks] = build_afrag(hrow, ks, hi);

        v8f gi[3] = {}, gh[3] = {};
#pragma unroll
        for (int g = 0; g < 3; ++g) {
            const int n = g * H_ + colBase + li;
#pragma unroll
            for (int ks = 0; ks < 4; ++ks) {
                const int kb = ks * 32 + hi * 8;
                v16h Bi = load_bfrag(WihL, n, kb);       // LDS, loop-invariant
                gi[g] = __builtin_amdgcn_wmma_f32_16x16x32_f16(false, Ak[ks], false, Bi,
                                                               (short)0, gi[g], false, false);
                v16h Bh = load_bfrag(WhhL, n, kb);
                gh[g] = __builtin_amdgcn_wmma_f32_16x16x32_f16(false, Ah[ks], false, Bh,
                                                               (short)0, gh[g], false, false);
            }
        }
        // gates: per-lane VALU on matching C layouts; write h(t) to write buffer
#pragma unroll
        for (int i = 0; i < 8; ++i) {
            float r  = sigmoidf_(gi[0][i] + gh[0][i] + bs_r);
            float u  = sigmoidf_(gi[1][i] + gh[1][i] + bs_z);
            float nv = tanhf(gi[2][i] + bi_n + r * (gh[2][i] + bh_n));
            float u2 = attv[i] * u;
            float h  = hreg[i];
            float hy = u2 * h + (1.0f - u2) * nv;
            h = (t < lenr[i]) ? hy : h;                  // packed-sequence hold
            hreg[i] = h;
            hwr[(hi * 8 + i) * HROW_PITCH + colBase + li] = h;
        }
        __syncthreads();   // publish h(t); bounds wave skew to one step
    }
#pragma unroll
    for (int i = 0; i < 8; ++i)
        out[(rowBase + hi * 8 + i) * H_ + colBase + li] = hreg[i];
}

// ---- host launch -----------------------------------------------------------
extern "C" void kernel_launch(void* const* d_in, const int* in_sizes, int n_in,
                              void* d_out, int out_size, void* d_ws, size_t ws_size,
                              hipStream_t stream) {
    (void)in_sizes; (void)n_in; (void)out_size; (void)ws_size;
    const float* query = (const float*)d_in[0];
    const float* keys  = (const float*)d_in[1];
    const int*   klen  = (const int*)d_in[2];
    const float* W1    = (const float*)d_in[3];
    const float* b1    = (const float*)d_in[4];
    const float* W2    = (const float*)d_in[5];
    const float* b2    = (const float*)d_in[6];
    const float* W3    = (const float*)d_in[7];
    const float* b3    = (const float*)d_in[8];
    const float* W_ih  = (const float*)d_in[9];
    const float* b_ih  = (const float*)d_in[10];
    const float* W_hh  = (const float*)d_in[11];
    const float* b_hh  = (const float*)d_in[12];

    char* ws = (char*)d_ws;
    float*    qrow   = (float*)(ws);                    // 2048*64*4    = 524288
    float*    scores = (float*)(ws + 524288);           // 2048*200*4   = 1638400
    _Float16* Wk16   = (_Float16*)(ws + 2162688);       // 64*128*2     = 16384
    _Float16* Wp16   = (_Float16*)(ws + 2179072);       // 64*128*2     = 16384
    _Float16* Wih16  = (_Float16*)(ws + 2195456);       // 384*128*2    = 98304
    _Float16* Whh16  = (_Float16*)(ws + 2293760);       // 384*128*2    = 98304

    prep_qrow_kernel<<<(B_ * A1_ + 255) / 256, 256, 0, stream>>>(query, W1, b1, qrow);
    prep_w16_kernel<<<(3 * H_ * H_ + 255) / 256, 256, 0, stream>>>(
        W1, W_ih, W_hh, Wk16, Wp16, Wih16, Whh16);
    attn_scores_kernel<<<(B_ * T_) / 64, 128, 0, stream>>>(
        query, keys, klen, qrow, Wk16, Wp16, W2, b2, W3, b3, scores);
    augru_kernel<<<B_ / 16, 256, AUGRU_LDS_BYTES, stream>>>(
        keys, klen, scores, Wih16, Whh16, b_ih, b_hh, (float*)d_out);
}
